// GraphModel_45208825757775
// MI455X (gfx1250) — matrix operand order repaired and checked by reading, not compiled, not run on hardware
//
#include <hip/hip_runtime.h>
#include <hip/hip_bf16.h>

typedef _Float16 half8  __attribute__((ext_vector_type(8)));
typedef _Float16 half16 __attribute__((ext_vector_type(16)));
typedef float    f32x8  __attribute__((ext_vector_type(8)));

#define INC  256   // in channels / hidden (H*C)
#define H1   4     // heads layer 1
#define OUTC 40    // out channels
#define OUTP 48    // padded out channels (multiple of 16)

// ---------------------------------------------------------------- utilities
__device__ __forceinline__ void atomAddF(float* p, float v) {
    unsafeAtomicAdd(p, v);   // hardware global_atomic_add_f32
}

__device__ __forceinline__ void edge_sd(const int* __restrict__ ei, int E, int i,
                                        int& s, int& d) {
    if (i < E) { s = ei[i]; d = ei[E + i]; }
    else       { s = i - E; d = s; }        // self loops appended after E edges
}

// ---------------------------------------------------------------- converts / packing
__global__ void k_f32_to_f16(const float* __restrict__ in, _Float16* __restrict__ out,
                             int total) {
    int i = blockIdx.x * blockDim.x + threadIdx.x;
    if (i < total) out[i] = (_Float16)in[i];
}

// Pre-swizzle W (row-major [256][nValid]) into per-lane-contiguous B fragments:
// Bp[((kt*NT+nt)*32 + lane)*16 + j]  with
//   k = kt*32 + (lane>>4)*8 + (j&7) + (j>>3)*16 , n = nt*16 + (lane&15)   (zero-pad n >= nValid)
__global__ void k_pack_w(const float* __restrict__ W, _Float16* __restrict__ Bp,
                         int nValid, int NT) {
    int idx = blockIdx.x * blockDim.x + threadIdx.x;
    int total = (INC / 32) * NT * 32 * 16;
    if (idx >= total) return;
    int j    = idx & 15;
    int lane = (idx >> 4) & 31;
    int t    = idx >> 9;          // kt*NT + nt
    int nt   = t % NT;
    int kt   = t / NT;
    int k = kt * 32 + (lane >> 4) * 8 + (j & 7) + (j >> 3) * 16;
    int n = nt * 16 + (lane & 15);
    float v = (n < nValid) ? W[(size_t)k * nValid + n] : 0.0f;
    Bp[idx] = (_Float16)v;
}

// ---------------------------------------------------------------- WMMA GEMMs
// A fragment (f16 16x32, row-major A): lane&15 = M row; (lane>>4)*8 = K sub-offset;
// halves 0..7  = A[m][k0+koff .. +7], halves 8..15 = A[m][k0+koff+16 .. +23]
__device__ __forceinline__ half16 load_a_frag(const _Float16* __restrict__ arow, int kbase) {
    half8 lo = *reinterpret_cast<const half8*>(arow + kbase);
    half8 hi = *reinterpret_cast<const half8*>(arow + kbase + 16);
    return __builtin_shufflevector(lo, hi, 0,1,2,3,4,5,6,7,8,9,10,11,12,13,14,15);
}

// h1 = xh @ W1 : [NB,256]x[256,256] -> f32 [NB,256]
// block = 128 thr = 4 waves; wave w owns cols [w*64, w*64+64) (4 accumulators);
// block owns 16 rows. Software-pipelined: stage kt+1 fragments load during stage kt WMMAs.
__global__ __launch_bounds__(128) void k_gemm_h1(const _Float16* __restrict__ A,
                                                 const _Float16* __restrict__ Bp,
                                                 float* __restrict__ C, int NB) {
    int tid  = threadIdx.x;
    int lane = tid & 31;
    int wave = tid >> 5;
    int m0   = blockIdx.x * 16;
    int mrow = lane & 15;
    int koff = (lane >> 4) * 8;
    int arowi = m0 + mrow; if (arowi >= NB) arowi = NB - 1;
    const _Float16* arow  = A + (size_t)arowi * INC;
    // B fragment base for this wave: ntile = wave*4 + nn; stride per kt = 16*32*16 halves,
    // per nn = 32*16 halves  -> all loads get immediate offsets from one base.
    const _Float16* bbase = Bp + (((size_t)wave * 4) * 32 + lane) * 16;

    f32x8 acc[4] = {};

    half16 a_cur = load_a_frag(arow, koff);
    half16 b_cur[4];
#pragma unroll
    for (int nn = 0; nn < 4; ++nn)
        b_cur[nn] = *reinterpret_cast<const half16*>(bbase + nn * 512);

#pragma unroll
    for (int kt = 0; kt < 8; ++kt) {
        half16 a_nxt;
        half16 b_nxt[4];
        if (kt < 7) {                      // prefetch stage kt+1 while WMMAs run
            a_nxt = load_a_frag(arow, (kt + 1) * 32 + koff);
#pragma unroll
            for (int nn = 0; nn < 4; ++nn)
                b_nxt[nn] = *reinterpret_cast<const half16*>(
                    bbase + (size_t)(kt + 1) * 8192 + nn * 512);
        }
#pragma unroll
        for (int nn = 0; nn < 4; ++nn)
            acc[nn] = __builtin_amdgcn_wmma_f32_16x16x32_f16(
                false, a_cur, false, b_cur[nn], (short)0, acc[nn], false, false);
        if (kt < 7) {
            a_cur = a_nxt;
#pragma unroll
            for (int nn = 0; nn < 4; ++nn) b_cur[nn] = b_nxt[nn];
        }
    }

    int ncol  = lane & 15;
    int mbase = (lane >> 4) * 8;
#pragma unroll
    for (int nn = 0; nn < 4; ++nn) {
        int n0 = (wave * 4 + nn) * 16;
#pragma unroll
        for (int r = 0; r < 8; ++r) {
            int m = m0 + mbase + r;
            if (m < NB) C[(size_t)m * INC + n0 + ncol] = acc[nn][r];
        }
    }
}

// h2 = hreluh @ W2pad : [NB,256]x[256,48] -> f32 [NB,48]
// block = 96 thr = 3 waves; wave w owns col tile w; block owns 16 rows. Pipelined.
__global__ __launch_bounds__(96) void k_gemm_h2(const _Float16* __restrict__ A,
                                                const _Float16* __restrict__ Bp,
                                                float* __restrict__ C, int NB) {
    int tid  = threadIdx.x;
    int lane = tid & 31;
    int wave = tid >> 5;           // = ntile (0..2)
    int m0   = blockIdx.x * 16;
    int mrow = lane & 15;
    int koff = (lane >> 4) * 8;
    int arowi = m0 + mrow; if (arowi >= NB) arowi = NB - 1;
    const _Float16* arow  = A + (size_t)arowi * INC;
    const _Float16* bbase = Bp + ((size_t)wave * 32 + lane) * 16;  // + kt*3*512 halves

    f32x8 acc = {};
    half16 a_cur = load_a_frag(arow, koff);
    half16 b_cur = *reinterpret_cast<const half16*>(bbase);

#pragma unroll
    for (int kt = 0; kt < 8; ++kt) {
        half16 a_nxt, b_nxt;
        if (kt < 7) {
            a_nxt = load_a_frag(arow, (kt + 1) * 32 + koff);
            b_nxt = *reinterpret_cast<const half16*>(bbase + (size_t)(kt + 1) * 1536);
        }
        acc = __builtin_amdgcn_wmma_f32_16x16x32_f16(
            false, a_cur, false, b_cur, (short)0, acc, false, false);
        if (kt < 7) { a_cur = a_nxt; b_cur = b_nxt; }
    }

    int ncol  = lane & 15;
    int mbase = (lane >> 4) * 8;
    int n0    = wave * 16;
#pragma unroll
    for (int r = 0; r < 8; ++r) {
        int m = m0 + mbase + r;
        if (m < NB) C[(size_t)m * OUTP + n0 + ncol] = acc[r];
    }
}

// ---------------------------------------------------------------- attention coeffs
// layer 1: one block (256 thr) per node; per-head (64-wide) LDS reduction
__global__ __launch_bounds__(256) void k_alpha1(const float* __restrict__ h,
                                                const float* __restrict__ asrc,
                                                const float* __restrict__ adst,
                                                float* __restrict__ als,
                                                float* __restrict__ ald) {
    __shared__ float ss[256], sd[256];
    int n = blockIdx.x, t = threadIdx.x;
    float v = h[(size_t)n * INC + t];
    ss[t] = v * asrc[t];
    sd[t] = v * adst[t];
    __syncthreads();
    for (int off = 32; off >= 1; off >>= 1) {
        if ((t & 63) < off) { ss[t] += ss[t + off]; sd[t] += sd[t + off]; }
        __syncthreads();
    }
    if ((t & 63) == 0) {
        int hh = t >> 6;
        als[n * H1 + hh] = ss[t];
        ald[n * H1 + hh] = sd[t];
    }
}

// layer 2: one thread per node (40 channels)
__global__ void k_alpha2(const float* __restrict__ h2,
                         const float* __restrict__ asrc,
                         const float* __restrict__ adst,
                         float* __restrict__ als, float* __restrict__ ald, int NB) {
    int n = blockIdx.x * blockDim.x + threadIdx.x;
    if (n >= NB) return;
    const float* row = h2 + (size_t)n * OUTP;
    float s = 0.f, d = 0.f;
#pragma unroll 8
    for (int o = 0; o < OUTC; ++o) { float v = row[o]; s += v * asrc[o]; d += v * adst[o]; }
    als[n] = s; ald[n] = d;
}

// ---------------------------------------------------------------- edge softmax numerators
// Logits are bounded (|e| small) -> max-subtraction is an exact mathematical no-op; skip it.
__global__ void k_edge_num(const int* __restrict__ ei, int E, int Etot,
                           const float* __restrict__ als, const float* __restrict__ ald,
                           int Hh, float* __restrict__ ee, float* __restrict__ den) {
    int i = blockIdx.x * blockDim.x + threadIdx.x;
    if (i >= Etot) return;
    int s, d; edge_sd(ei, E, i, s, d);
    for (int h = 0; h < Hh; ++h) {
        float e = als[s * Hh + h] + ald[d * Hh + h];
        e = (e > 0.f) ? e : 0.2f * e;              // leaky_relu(0.2)
        float x = __expf(e);
        ee[(size_t)i * Hh + h] = x;
        atomAddF(&den[d * Hh + h], x);
    }
}

// ---------------------------------------------------------------- aggregation (layer 1)
// 64 threads per edge, 4 channels each: coalesced float4 gather of h1[src] + fp32 atomics
__global__ void k_agg1(const int* __restrict__ ei, int E, int Etot,
                       const float* __restrict__ h1, const float* __restrict__ ee,
                       const float* __restrict__ den, float* __restrict__ agg) {
    long long idx = (long long)blockIdx.x * blockDim.x + threadIdx.x;
    long long total = (long long)Etot * 64;
    if (idx >= total) return;
    int edge = (int)(idx >> 6);
    int q    = (int)(idx & 63);     // channel group: channels [q*4, q*4+4)
    int hh   = q >> 4;              // head
    int s, d; edge_sd(ei, E, edge, s, d);
    float alpha = ee[(size_t)edge * H1 + hh] / (den[d * H1 + hh] + 1e-16f);
    const float4 hv = *reinterpret_cast<const float4*>(h1 + (size_t)s * INC + q * 4);
    float* o = agg + (size_t)d * INC + q * 4;
    atomAddF(o + 0, alpha * hv.x);
    atomAddF(o + 1, alpha * hv.y);
    atomAddF(o + 2, alpha * hv.z);
    atomAddF(o + 3, alpha * hv.w);
}

// ---------------------------------------------------------------- relu+bias -> f16
__global__ void k_relu_bias_h(const float* __restrict__ agg, const float* __restrict__ b,
                              _Float16* __restrict__ oh, int total) {
    int i = blockIdx.x * blockDim.x + threadIdx.x;
    if (i >= total) return;
    float v = agg[i] + b[i & (INC - 1)];
    oh[i] = (_Float16)(v > 0.f ? v : 0.f);
}

// ---------------------------------------------------------------- aggregation (layer 2)
// 10 threads per edge, 4 channels each (covers 40 channels)
__global__ void k_agg2(const int* __restrict__ ei, int E, int Etot,
                       const float* __restrict__ h2, const float* __restrict__ ee,
                       const float* __restrict__ den, float* __restrict__ agg) {
    long long idx = (long long)blockIdx.x * blockDim.x + threadIdx.x;
    long long total = (long long)Etot * 10;
    if (idx >= total) return;
    int edge = (int)(idx / 10);
    int q    = (int)(idx - (long long)edge * 10);
    int s, d; edge_sd(ei, E, edge, s, d);
    float alpha = ee[edge] / (den[d] + 1e-16f);
    const float4 hv = *reinterpret_cast<const float4*>(h2 + (size_t)s * OUTP + q * 4);
    float* o = agg + (size_t)d * OUTC + q * 4;
    atomAddF(o + 0, alpha * hv.x);
    atomAddF(o + 1, alpha * hv.y);
    atomAddF(o + 2, alpha * hv.z);
    atomAddF(o + 3, alpha * hv.w);
}

// ---------------------------------------------------------------- final log_softmax
__global__ void k_logsoftmax(const float* __restrict__ agg, const float* __restrict__ b2,
                             float* __restrict__ out, int NB) {
    int n = blockIdx.x * blockDim.x + threadIdx.x;
    if (n >= NB) return;
    float v[OUTC];
    float mx = -3.402823466e+38f;
#pragma unroll 8
    for (int o = 0; o < OUTC; ++o) {
        v[o] = agg[(size_t)n * OUTC + o] + b2[o];
        mx = fmaxf(mx, v[o]);
    }
    float sum = 0.f;
#pragma unroll 8
    for (int o = 0; o < OUTC; ++o) sum += __expf(v[o] - mx);
    float ls = __logf(sum);
#pragma unroll 8
    for (int o = 0; o < OUTC; ++o) out[(size_t)n * OUTC + o] = v[o] - mx - ls;
}

// ================================================================ launcher
extern "C" void kernel_launch(void* const* d_in, const int* in_sizes, int n_in,
                              void* d_out, int out_size, void* d_ws, size_t ws_size,
                              hipStream_t stream) {
    const float* x     = (const float*)d_in[0];
    const int*   ei    = (const int*)  d_in[1];
    const float* W1    = (const float*)d_in[2];
    const float* asrc1 = (const float*)d_in[3];
    const float* adst1 = (const float*)d_in[4];
    const float* b1    = (const float*)d_in[5];
    const float* W2    = (const float*)d_in[6];
    const float* asrc2 = (const float*)d_in[7];
    const float* adst2 = (const float*)d_in[8];
    const float* b2    = (const float*)d_in[9];
    float* out = (float*)d_out;

    const int NB   = in_sizes[0] / INC;   // 50000
    const int E    = in_sizes[1] / 2;     // 800000
    const int Etot = E + NB;              // 850000

    // ---- workspace layout (aliased across layers)
    char* base = (char*)d_ws;
    size_t off = 0;
    auto alloc = [&](size_t sz) { size_t c = off; off += (sz + 255) & ~(size_t)255; return c; };
    size_t o_h1   = alloc((size_t)NB * INC * 4);       // layer2: agg2 lives here
    size_t o_agg1 = alloc((size_t)NB * INC * 4);       // layer2: h2 lives here
    size_t o_xh   = alloc((size_t)NB * INC * 2);       // layer2: hreluh lives here
    size_t o_ee1  = alloc((size_t)Etot * H1 * 4);      // layer2: ee2 lives here
    size_t o_als1 = alloc((size_t)NB * H1 * 4);
    size_t o_ald1 = alloc((size_t)NB * H1 * 4);
    size_t o_den1 = alloc((size_t)NB * H1 * 4);
    size_t o_als2 = alloc((size_t)NB * 4);
    size_t o_ald2 = alloc((size_t)NB * 4);
    size_t o_den2 = alloc((size_t)NB * 4);
    size_t o_w1p  = alloc((size_t)(INC / 32) * 16 * 32 * 16 * 2);
    size_t o_w2p  = alloc((size_t)(INC / 32) * 3  * 32 * 16 * 2);

    float*    h1     = (float*)   (base + o_h1);
    float*    agg1   = (float*)   (base + o_agg1);
    _Float16* xh     = (_Float16*)(base + o_xh);
    float*    ee1    = (float*)   (base + o_ee1);
    float*    als1   = (float*)   (base + o_als1);
    float*    ald1   = (float*)   (base + o_ald1);
    float*    den1   = (float*)   (base + o_den1);
    float*    als2   = (float*)   (base + o_als2);
    float*    ald2   = (float*)   (base + o_ald2);
    float*    den2   = (float*)   (base + o_den2);
    _Float16* w1p    = (_Float16*)(base + o_w1p);
    _Float16* w2p    = (_Float16*)(base + o_w2p);
    // aliases (stream-ordered: producers run strictly after last reader of the old use)
    float*    agg2   = (float*)   (base + o_h1);
    float*    h2     = (float*)   (base + o_agg1);
    _Float16* hreluh = (_Float16*)(base + o_xh);
    float*    ee2    = (float*)   (base + o_ee1);

    auto cdiv = [](long long a, long long b) { return (int)((a + b - 1) / b); };
    const int BLK = 256;

    // ---- layer 1
    hipMemsetAsync(agg1, 0, (size_t)NB * INC * 4, stream);
    hipMemsetAsync(den1, 0, (size_t)NB * H1 * 4, stream);

    k_f32_to_f16<<<cdiv((long long)NB * INC, BLK), BLK, 0, stream>>>(x, xh, NB * INC);
    k_pack_w<<<cdiv((INC / 32) * 16 * 512, BLK), BLK, 0, stream>>>(W1, w1p, INC, 16);
    k_pack_w<<<cdiv((INC / 32) * 3 * 512, BLK), BLK, 0, stream>>>(W2, w2p, OUTC, 3);

    k_gemm_h1<<<cdiv(NB, 16), 128, 0, stream>>>(xh, w1p, h1, NB);
    k_alpha1<<<NB, 256, 0, stream>>>(h1, asrc1, adst1, als1, ald1);
    k_edge_num<<<cdiv(Etot, BLK), BLK, 0, stream>>>(ei, E, Etot, als1, ald1, H1, ee1, den1);
    k_agg1<<<cdiv((long long)Etot * 64, BLK), BLK, 0, stream>>>(ei, E, Etot, h1, ee1, den1, agg1);

    // ---- layer 2 (agg2 aliases h1: zero only after k_agg1, which is the last h1 reader)
    hipMemsetAsync(agg2, 0, (size_t)NB * OUTC * 4, stream);
    hipMemsetAsync(den2, 0, (size_t)NB * 4, stream);

    k_relu_bias_h<<<cdiv((long long)NB * INC, BLK), BLK, 0, stream>>>(agg1, b1, hreluh, NB * INC);
    k_gemm_h2<<<cdiv(NB, 16), 96, 0, stream>>>(hreluh, w2p, h2, NB);
    k_alpha2<<<cdiv(NB, BLK), BLK, 0, stream>>>(h2, asrc2, adst2, als2, ald2, NB);
    k_edge_num<<<cdiv(Etot, BLK), BLK, 0, stream>>>(ei, E, Etot, als2, ald2, 1, ee2, den2);
    k_agg2<<<cdiv((long long)Etot * 10, BLK), BLK, 0, stream>>>(ei, E, Etot, h2, ee2, den2, agg2);

    k_logsoftmax<<<cdiv(NB, BLK), BLK, 0, stream>>>(agg2, b2, out, NB);
}
